// RnnSampler_20658792694412
// MI455X (gfx1250) — matrix operand order repaired and checked by reading, not compiled, not run on hardware
//
#include <hip/hip_runtime.h>

// ---------------------------------------------------------------------------
// RNN sampler for MI455X (gfx1250): bf16 WMMA GEMMs, async-LDS B staging,
// fused Gumbel-argmax sampling.  E=256, V=32000, N=256 samples, 128 steps.
// ---------------------------------------------------------------------------

typedef __bf16 bf16_t;
typedef __attribute__((ext_vector_type(8)))  __bf16 v8bf;
typedef __attribute__((ext_vector_type(16))) __bf16 v16bf;
typedef __attribute__((ext_vector_type(8)))  float  v8f;

#define EDIM   256
#define VOCAB  32000
#define NSAMP  256
#define NSTEPS 128

// LDS B-chunk layout: 128 columns x 32 K, row stride padded to 40 elements
// (80 B) -> 16-byte aligned and conflict-free across the 16 fragment lanes.
#define SB_STRIDE 40

#if __has_builtin(__builtin_amdgcn_global_load_async_to_lds_b128)
#define HAVE_ASYNC_LDS 1
#else
#define HAVE_ASYNC_LDS 0
#endif

#if HAVE_ASYNC_LDS
// Builtin signature (from clang diagnostic): (int4 AS1*, int4 AS3*, imm, imm)
typedef int v4i_vec __attribute__((vector_size(16)));
typedef __attribute__((address_space(1))) v4i_vec* as1_v4i_p;
typedef __attribute__((address_space(3))) v4i_vec* as3_v4i_p;
#if __has_builtin(__builtin_amdgcn_s_wait_asynccnt)
#define WAIT_ASYNC(n) __builtin_amdgcn_s_wait_asynccnt(n)
#else
#define WAIT_ASYNC(n) asm volatile("s_wait_asynccnt %0" :: "n"(n) : "memory")
#endif
#endif

// ---- WMMA fragment loaders (contiguous-K order shared by A and B, wave32) -------------------
// A (16x32 bf16, MxK): lane L<16 holds row M=L, K = k0+0..15;
// lane L>=16 holds row M=L-16, K = k0+16..31.
__device__ __forceinline__ v16bf load_frag_a(const bf16_t* __restrict__ A,
                                             int lda, int m0, int k0, int lane) {
    int row = m0 + (lane & 15);
    int kb  = k0 + ((lane >> 4) << 4);
    const bf16_t* p = A + (size_t)row * lda + kb;
    v8bf lo = *(const v8bf*)(p);
    v8bf hi = *(const v8bf*)(p + 8);
    return __builtin_shufflevector(lo, hi, 0,1,2,3,4,5,6,7,8,9,10,11,12,13,14,15);
}

// B (32x16 bf16, KxN) from row-major weight W[n][k] (B = W^T):
// lane L holds col N=L&15, K = k0 + (L<16 ? 0..15 : 16..31)  -> 16 contiguous.
__device__ __forceinline__ v16bf load_frag_b(const bf16_t* __restrict__ W,
                                             int ldw, int n0, int k0, int lane) {
    int col = n0 + (lane & 15);
    int kb  = k0 + ((lane >> 4) << 4);
    const bf16_t* p = W + (size_t)col * ldw + kb;
    v8bf lo = *(const v8bf*)(p);
    v8bf hi = *(const v8bf*)(p + 8);
    return __builtin_shufflevector(lo, hi, 0,1,2,3,4,5,6,7,8,9,10,11,12,13,14,15);
}

// Same B fragment but from the staged LDS chunk (block-relative column).
__device__ __forceinline__ v16bf load_frag_b_lds(const bf16_t* buf, int j, int lane) {
    const bf16_t* p = buf + (size_t)(16 * j + (lane & 15)) * SB_STRIDE + ((lane >> 4) << 4);
    v8bf lo = *(const v8bf*)(p);
    v8bf hi = *(const v8bf*)(p + 8);
    return __builtin_shufflevector(lo, hi, 0,1,2,3,4,5,6,7,8,9,10,11,12,13,14,15);
}

// Stage one 128-col x 32-K bf16 chunk of W2 into LDS.
// 256 threads: thread t copies 32 B (col = t>>1, half = t&1).
// Async path: 2 ASYNC instructions per wave (ASYNCcnt += 2).
__device__ __forceinline__ void fill_chunk(bf16_t* dst, const bf16_t* __restrict__ W,
                                           int n0, int k0, int tid) {
    int col = tid >> 1, half = tid & 1;
    const bf16_t* g = W + (size_t)(n0 + col) * (2 * EDIM) + k0 + half * 16;
    bf16_t* l = dst + (size_t)col * SB_STRIDE + half * 16;
#if HAVE_ASYNC_LDS
    __builtin_amdgcn_global_load_async_to_lds_b128((as1_v4i_p)(void*)g,
                                                   (as3_v4i_p)(void*)l, 0, 0);
    __builtin_amdgcn_global_load_async_to_lds_b128((as1_v4i_p)(void*)(g + 8),
                                                   (as3_v4i_p)(void*)(l + 8), 0, 0);
#else
    *(uint4*)l       = *(const uint4*)g;
    *(uint4*)(l + 8) = *(const uint4*)(g + 8);
#endif
}

__device__ __forceinline__ float sigmoidf_(float x) {
    return 1.0f / (1.0f + __expf(-x));
}

// Monotonic float -> uint mapping (order-preserving)
__device__ __forceinline__ unsigned fmono(float f) {
    unsigned b = __float_as_uint(f);
    return (b & 0x80000000u) ? ~b : (b | 0x80000000u);
}

// Deterministic per-(step,row,col) Gumbel noise via 32-bit hash
__device__ __forceinline__ float gumbel_noise(unsigned s, unsigned row, unsigned col) {
    unsigned x = s * 0x9E3779B9u ^ row * 0x85EBCA6Bu ^ col * 0xC2B2AE35u;
    x ^= x >> 16; x *= 0x7FEB352Du;
    x ^= x >> 15; x *= 0x846CA68Bu;
    x ^= x >> 16;
    float u = (float)(x >> 8) * (1.0f / 16777216.0f);
    u = fminf(fmaxf(u, 1e-7f), 1.0f - 1e-7f);
    return -__logf(-__logf(u));
}

// ---------------------------------------------------------------------------
// Prep kernels
// ---------------------------------------------------------------------------
__global__ void cvt_f32_bf16(const float* __restrict__ src, bf16_t* __restrict__ dst, int n) {
    for (int i = blockIdx.x * blockDim.x + threadIdx.x; i < n; i += gridDim.x * blockDim.x)
        dst[i] = (bf16_t)src[i];
}

__global__ void init_state(const float* __restrict__ init_h,
                           float* __restrict__ c, bf16_t* __restrict__ hbf) {
    int idx = blockIdx.x * blockDim.x + threadIdx.x;   // 0 .. N*E-1
    float v = init_h[idx & (EDIM - 1)];
    c[idx]   = v;
    hbf[idx] = (bf16_t)v;
}

// ---------------------------------------------------------------------------
// Step kernel 1: a = relu(h @ W1^T + b1) -> bf16 [N, 2E]; also zero argmax keys
// grid (2, 8), block 256 (8 waves; wave = 16x64 strip)
// ---------------------------------------------------------------------------
__global__ void mlp1_kernel(const bf16_t* __restrict__ hbf, const bf16_t* __restrict__ W1b,
                            const float* __restrict__ b1, bf16_t* __restrict__ abf,
                            unsigned long long* __restrict__ keys) {
    int tid = threadIdx.x;
    if (blockIdx.x == 0 && blockIdx.y == 0) keys[tid] = 0ull;  // reset for this step

    int wave = tid >> 5, lane = tid & 31;
    int m0 = blockIdx.x * 128 + wave * 16;
    int n0 = blockIdx.y * 64;
    v8f acc[4] = {};
    for (int k0 = 0; k0 < EDIM; k0 += 32) {
        v16bf a = load_frag_a(hbf, EDIM, m0, k0, lane);
        v16bf bf[4];
#pragma unroll
        for (int j = 0; j < 4; ++j) bf[j] = load_frag_b(W1b, EDIM, n0 + 16 * j, k0, lane);
#pragma unroll
        for (int j = 0; j < 4; ++j)
            acc[j] = __builtin_amdgcn_wmma_f32_16x16x32_bf16(
                false, a, false, bf[j], (short)0, acc[j], false, false);
    }
    int rbase = m0 + ((lane >> 4) << 3);
#pragma unroll
    for (int j = 0; j < 4; ++j) {
        int col = n0 + 16 * j + (lane & 15);
        float bias = b1[col];
#pragma unroll
        for (int r = 0; r < 8; ++r) {
            float v = acc[j][r] + bias;
            abf[(size_t)(rbase + r) * (2 * EDIM) + col] = (bf16_t)(v > 0.0f ? v : 0.0f);
        }
    }
}

// ---------------------------------------------------------------------------
// Step kernel 2 (dominant): logits = a @ W2^T + b2, fused Gumbel-argmax.
// grid (2, 250), block 256. Wave: 16(M) x 128(N), K = 512.
// W2 chunk (128 cols x 32 K) staged in LDS via async-to-LDS, double-buffered;
// all 8 waves consume the shared chunk. W2_bf (32.8 MB) is L2-resident.
// ---------------------------------------------------------------------------
__global__ void logits_kernel(const bf16_t* __restrict__ abf, const bf16_t* __restrict__ W2b,
                              const float* __restrict__ b2,
                              unsigned long long* __restrict__ keys, int step) {
    __shared__ bf16_t sB[2][128 * SB_STRIDE];

    int tid = threadIdx.x, wave = tid >> 5, lane = tid & 31;
    int m0 = blockIdx.x * 128 + wave * 16;
    int n0 = blockIdx.y * 128;

    v8f acc[8] = {};
    fill_chunk(sB[0], W2b, n0, 0, tid);
    int p = 0;
    for (int k0 = 0; k0 < 2 * EDIM; k0 += 32) {
        bool more = (k0 + 32) < (2 * EDIM);
        if (more) fill_chunk(sB[p ^ 1], W2b, n0, k0 + 32, tid);
#if HAVE_ASYNC_LDS
        if (more) { WAIT_ASYNC(2); } else { WAIT_ASYNC(0); }
#endif
        __syncthreads();

        // prefetch next A chunk (gfx1250 global_prefetch_b8)
        __builtin_prefetch((const void*)(abf + (size_t)(m0 + (lane & 15)) * (2 * EDIM) + k0 + 32), 0, 0);
        v16bf a = load_frag_a(abf, 2 * EDIM, m0, k0, lane);
        v16bf bf[8];
#pragma unroll
        for (int j = 0; j < 8; ++j) bf[j] = load_frag_b_lds(sB[p], j, lane);
#pragma unroll
        for (int j = 0; j < 8; ++j)
            acc[j] = __builtin_amdgcn_wmma_f32_16x16x32_bf16(
                false, a, false, bf[j], (short)0, acc[j], false, false);

        __syncthreads();           // everyone done reading sB[p] before refill
        p ^= 1;
    }

    // Epilogue: bias + Gumbel noise, per-row argmax over this wave's 128 cols,
    // then one u64 atomicMax per row with packed {monotonic value, col} key.
    float bias[8];
#pragma unroll
    for (int j = 0; j < 8; ++j) bias[j] = b2[n0 + 16 * j + (lane & 15)];

    int rbase = m0 + ((lane >> 4) << 3);
#pragma unroll
    for (int r = 0; r < 8; ++r) {
        unsigned row = (unsigned)(rbase + r);
        float best = -3.4e38f;
        unsigned bcol = 0;
#pragma unroll
        for (int j = 0; j < 8; ++j) {
            unsigned col = (unsigned)(n0 + 16 * j + (lane & 15));
            float v = acc[j][r] + bias[j] + gumbel_noise((unsigned)step, row, col);
            if (v > best) { best = v; bcol = col; }
        }
        // reduce across the 16 lanes that share this row (lanes xor 1,2,4,8)
#pragma unroll
        for (int m = 1; m < 16; m <<= 1) {
            float    ov = __shfl_xor(best, m);
            unsigned oc = (unsigned)__shfl_xor((int)bcol, m);
            if (ov > best || (ov == best && oc > bcol)) { best = ov; bcol = oc; }
        }
        if ((lane & 15) == 0) {
            unsigned long long key = ((unsigned long long)fmono(best) << 32) | (unsigned long long)bcol;
            atomicMax(keys + row, key);
        }
    }
}

// ---------------------------------------------------------------------------
// Step kernel 3: finalize sample, write output token, gather embedding row.
// grid NSAMP blocks x 64 threads (row copy: 64 x 8B = 512B = 256 bf16)
// ---------------------------------------------------------------------------
__global__ void sample_gather_kernel(const unsigned long long* __restrict__ keys,
                                     const bf16_t* __restrict__ embb,
                                     bf16_t* __restrict__ xbf,
                                     float* __restrict__ out, int step) {
    int n = blockIdx.x, tid = threadIdx.x;
    unsigned tok = (unsigned)(keys[n] & 0xFFFFFFFFull);
    if (tok >= (unsigned)VOCAB) tok = 0;
    if (tid == 0) out[(size_t)n * NSTEPS + step] = (float)tok;
    const uint2* src = (const uint2*)(embb + (size_t)tok * EDIM);
    uint2*       dst = (uint2*)(xbf + (size_t)n * EDIM);
    dst[tid] = src[tid];
}

// ---------------------------------------------------------------------------
// Step kernel 4: gates = x @ Wih^T + h @ Whh^T + b_ih + b_hh   [N, 4E] f32
// grid (2, 16), block 256. Wave: 16x64, two K=256 accumulation passes.
// ---------------------------------------------------------------------------
__global__ void gates_kernel(const bf16_t* __restrict__ xbf, const bf16_t* __restrict__ hbf,
                             const bf16_t* __restrict__ Wihb, const bf16_t* __restrict__ Whhb,
                             const float* __restrict__ b_ih, const float* __restrict__ b_hh,
                             float* __restrict__ gates) {
    int tid = threadIdx.x, wave = tid >> 5, lane = tid & 31;
    int m0 = blockIdx.x * 128 + wave * 16;
    int n0 = blockIdx.y * 64;
    v8f acc[4] = {};
    for (int k0 = 0; k0 < EDIM; k0 += 32) {
        v16bf a = load_frag_a(xbf, EDIM, m0, k0, lane);
        v16bf bf[4];
#pragma unroll
        for (int j = 0; j < 4; ++j) bf[j] = load_frag_b(Wihb, EDIM, n0 + 16 * j, k0, lane);
#pragma unroll
        for (int j = 0; j < 4; ++j)
            acc[j] = __builtin_amdgcn_wmma_f32_16x16x32_bf16(
                false, a, false, bf[j], (short)0, acc[j], false, false);
    }
    for (int k0 = 0; k0 < EDIM; k0 += 32) {
        v16bf a = load_frag_a(hbf, EDIM, m0, k0, lane);
        v16bf bf[4];
#pragma unroll
        for (int j = 0; j < 4; ++j) bf[j] = load_frag_b(Whhb, EDIM, n0 + 16 * j, k0, lane);
#pragma unroll
        for (int j = 0; j < 4; ++j)
            acc[j] = __builtin_amdgcn_wmma_f32_16x16x32_bf16(
                false, a, false, bf[j], (short)0, acc[j], false, false);
    }
    int rbase = m0 + ((lane >> 4) << 3);
#pragma unroll
    for (int j = 0; j < 4; ++j) {
        int col = n0 + 16 * j + (lane & 15);
        float bias = b_ih[col] + b_hh[col];
#pragma unroll
        for (int r = 0; r < 8; ++r)
            gates[(size_t)(rbase + r) * (4 * EDIM) + col] = acc[j][r] + bias;
    }
}

// ---------------------------------------------------------------------------
// Step kernel 5: LSTM pointwise (PyTorch gate order i,f,g,o); h kept in bf16.
// ---------------------------------------------------------------------------
__global__ void lstm_kernel(const float* __restrict__ gates,
                            float* __restrict__ c, bf16_t* __restrict__ hbf) {
    int idx = blockIdx.x * blockDim.x + threadIdx.x;   // 0 .. N*E-1
    int n = idx >> 8, e = idx & (EDIM - 1);
    const float* g = gates + (size_t)n * (4 * EDIM);
    float ig = sigmoidf_(g[e]);
    float fg = sigmoidf_(g[EDIM + e]);
    float gg = tanhf(g[2 * EDIM + e]);
    float og = sigmoidf_(g[3 * EDIM + e]);
    float cn = fg * c[idx] + ig * gg;
    c[idx]   = cn;
    hbf[idx] = (bf16_t)(og * tanhf(cn));
}

// ---------------------------------------------------------------------------
// Host launcher (graph-capture safe: kernels only, all on `stream`)
// ---------------------------------------------------------------------------
extern "C" void kernel_launch(void* const* d_in, const int* in_sizes, int n_in,
                              void* d_out, int out_size, void* d_ws, size_t ws_size,
                              hipStream_t stream) {
    (void)in_sizes; (void)n_in; (void)out_size; (void)ws_size;

    const float* init_h      = (const float*)d_in[2];
    const float* token_embed = (const float*)d_in[3];
    const float* W_ih        = (const float*)d_in[4];
    const float* W_hh        = (const float*)d_in[5];
    const float* b_ih        = (const float*)d_in[6];
    const float* b_hh        = (const float*)d_in[7];
    const float* W1          = (const float*)d_in[8];
    const float* b1          = (const float*)d_in[9];
    const float* W2          = (const float*)d_in[10];
    const float* b2          = (const float*)d_in[11];
    float* out = (float*)d_out;

    // Workspace carve-up (256-B aligned)
    char* ws = (char*)d_ws;
    size_t off = 0;
    auto carve = [&](size_t bytes) -> void* {
        void* p = ws + off;
        off += (bytes + 255) & ~(size_t)255;
        return p;
    };
    bf16_t* W2b  = (bf16_t*)carve((size_t)VOCAB * 2 * EDIM * 2);   // 32.8 MB (L2-resident)
    bf16_t* embb = (bf16_t*)carve((size_t)VOCAB * EDIM * 2);       // 16.4 MB
    bf16_t* W1b  = (bf16_t*)carve((size_t)2 * EDIM * EDIM * 2);
    bf16_t* Wihb = (bf16_t*)carve((size_t)4 * EDIM * EDIM * 2);
    bf16_t* Whhb = (bf16_t*)carve((size_t)4 * EDIM * EDIM * 2);
    bf16_t* hbf  = (bf16_t*)carve((size_t)NSAMP * EDIM * 2);
    bf16_t* xbf  = (bf16_t*)carve((size_t)NSAMP * EDIM * 2);
    bf16_t* abf  = (bf16_t*)carve((size_t)NSAMP * 2 * EDIM * 2);
    float*  cbuf = (float*)carve((size_t)NSAMP * EDIM * 4);
    float*  gbuf = (float*)carve((size_t)NSAMP * 4 * EDIM * 4);
    unsigned long long* keys = (unsigned long long*)carve((size_t)NSAMP * 8);

    // One-time bf16 conversion of weights/embedding + state init
    cvt_f32_bf16<<<2048, 256, 0, stream>>>(W2,          W2b,  VOCAB * 2 * EDIM);
    cvt_f32_bf16<<<2048, 256, 0, stream>>>(token_embed, embb, VOCAB * EDIM);
    cvt_f32_bf16<<<256,  256, 0, stream>>>(W1,          W1b,  2 * EDIM * EDIM);
    cvt_f32_bf16<<<512,  256, 0, stream>>>(W_ih,        Wihb, 4 * EDIM * EDIM);
    cvt_f32_bf16<<<512,  256, 0, stream>>>(W_hh,        Whhb, 4 * EDIM * EDIM);
    init_state<<<NSAMP, EDIM, 0, stream>>>(init_h, cbuf, hbf);

    for (int t = 0; t < NSTEPS; ++t) {
        mlp1_kernel<<<dim3(2, 8),   256, 0, stream>>>(hbf, W1b, b1, abf, keys);
        logits_kernel<<<dim3(2, VOCAB / 128), 256, 0, stream>>>(abf, W2b, b2, keys, t);
        sample_gather_kernel<<<NSAMP, 64, 0, stream>>>(keys, embb, xbf, out, t);
        gates_kernel<<<dim3(2, 16), 256, 0, stream>>>(xbf, hbf, Wihb, Whhb, b_ih, b_hh, gbuf);
        lstm_kernel<<<NSAMP, EDIM, 0, stream>>>(gbuf, cbuf, hbf);
    }
}
